// vSU2_CNN_full_41979010351327
// MI455X (gfx1250) — matrix operand order repaired and checked by reading, not compiled, not run on hardware
//
#include <hip/hip_runtime.h>
#include <hip/hip_bf16.h>

typedef __attribute__((ext_vector_type(16))) _Float16 v16h;
typedef __attribute__((ext_vector_type(2)))  _Float16 v2h;
typedef __attribute__((ext_vector_type(8)))  float    v8f;
typedef __attribute__((ext_vector_type(8)))  unsigned v8u;

// |det(t_a,t_b,t_c)| for distinct vertices of the regular tetrahedron = 4/(3*sqrt(3))
#define TPV 0.769800358919501f

// TRIPLE_PRODUCT[a][b][c], flat index a*16+b*4+c. Fully antisymmetric.
__device__ __constant__ float g_TP[64] = {
  0.f, 0.f, 0.f, 0.f,   0.f, 0.f, TPV, -TPV,   0.f, -TPV, 0.f, TPV,   0.f, TPV, -TPV, 0.f,
  0.f, 0.f, -TPV, TPV,  0.f, 0.f, 0.f, 0.f,    TPV, 0.f, 0.f, -TPV,   -TPV, 0.f, TPV, 0.f,
  0.f, TPV, 0.f, -TPV,  -TPV, 0.f, 0.f, TPV,   0.f, 0.f, 0.f, 0.f,    TPV, -TPV, 0.f, 0.f,
  0.f, -TPV, TPV, 0.f,  TPV, 0.f, -TPV, 0.f,   -TPV, TPV, 0.f, 0.f,   0.f, 0.f, 0.f, 0.f
};

// Branchless ELU: exact for v>0 (exp(0)==1 exactly), hw v_exp_f32 for v<=0.
__device__ __forceinline__ float elu_f(float v) {
  return fmaxf(v, 0.0f) + (__expf(fminf(v, 0.0f)) - 1.0f);
}

// Pack two f32 -> one word of two f16 (lowers to v_cvt_pk_f16_f32).
__device__ __forceinline__ unsigned pk16(float a, float b) {
  v2h h;
  h.x = (_Float16)a;
  h.y = (_Float16)b;
  unsigned r;
  __builtin_memcpy(&r, &h, 4);
  return r;
}

// One wave handles one 16-output tile:
//   stage 1: lanes 0..19 each build one 6-channel y row, stored as 3 packed-f16 words in LDS
//   stage 2: one v_wmma_f32_16x16x32_f16: A = 16x30 sliding windows (K-padded to 32),
//            B = kernel_0 as 30x3 (N-padded to 16); then elu+bias and accumulate the mean.
// 16-bit A layout consumes adjacent even K-pairs per VGPR, so every A register is one
// aligned 32-bit LDS word: word(m, K) = 3*m + K/2.
__global__ __launch_bounds__(256) void su2cnn_main_kernel(
    const int*   __restrict__ x,
    const float* __restrict__ ikd,    // (3,5)
    const float* __restrict__ ikt,    // (3,5,5)
    const float* __restrict__ k0,     // (5,6,3) ; B[kk][f] = k0[3*kk+f]
    const float* __restrict__ bias,   // (3,)
    float*       __restrict__ partials,
    int n, int tiles, int iters, int wavesTotal)
{
  __shared__ unsigned shyw[8][60];   // per-wave 20 rows x 6 f16 channels = 60 words
  __shared__ float shTP[64];
  __shared__ float shikd[15];
  __shared__ float shanti[75];       // ikt - ikt^T (in j,k)
  __shared__ float shbias[3];
  __shared__ float shred[8][2][3];

  const int tid  = threadIdx.x;
  const int lane = tid & 31;
  const int wv   = tid >> 5;
  const int waveGlobal = blockIdx.x * 8 + wv;

  if (tid < 64) shTP[tid] = g_TP[tid];
  if (tid < 15) shikd[tid] = ikd[tid];
  if (tid < 75) {
    int f = tid / 25, r = tid % 25, j = r / 5, kq = r % 5;
    shanti[tid] = ikt[f*25 + j*5 + kq] - ikt[f*25 + kq*5 + j];
  }
  if (tid < 3) shbias[tid] = bias[tid];
  __syncthreads();

  const int  mrow = lane & 15;
  const bool hi   = lane >= 16;
  const int  kbA  = hi ? 8  : 0;   // 16-bit A layout: low lanes K{0-7,16-23}, high K{8-15,24-31}
  const int  kbB  = hi ? 16 : 0;   // unused: B is built with kbA, in the same K order as A

  // Bias hoisted into a register (no LDS traffic after the WMMA).
  const float bb = (mrow < 3) ? shbias[mrow] : 0.0f;

  // B operand (constant across tiles): B[K][ncol], K>=30 or ncol>=3 -> 0
  v16h Bv;
#pragma unroll
  for (int v = 0; v < 16; ++v) {
    int K = kbA + ((v < 8) ? v : v + 8);
    float bval = (K < 30 && mrow < 3) ? k0[3*K + mrow] : 0.0f;
    Bv[v] = (_Float16)bval;
  }

  float fsum = 0.0f;

  for (int it = 0; it < iters; ++it) {
    int t = it * wavesTotal + waveGlobal;
    bool active = t < tiles;   // wave-uniform -> EXEC stays all-ones around WMMA

    // ---- stage 1: build 20 y rows for this tile (packed f16 in LDS) ----
    if (active && lane < 20) {
      int g = t * 16 + lane;
      if (g >= n) g -= n;
      int xc = x[g];
      int xnv[5];
#pragma unroll
      for (int j = 0; j < 5; ++j) {
        int idx = g + 1 + j;
        if (idx >= n) idx -= n;
        xnv[j] = x[idx];
      }
      // prefetch next tile's x for this wave (speculative, global_prefetch_b8)
      {
        long long tn = (long long)t + wavesTotal;
        if (tn < tiles) __builtin_prefetch(x + tn * 16 + lane, 0, 0);
      }
      float ydot[3] = {0.f, 0.f, 0.f};
#pragma unroll
      for (int j = 0; j < 5; ++j) {
        float s = (xc == xnv[j]) ? 1.0f : -1.0f;
#pragma unroll
        for (int f = 0; f < 3; ++f) ydot[f] = fmaf(shikd[f*5 + j], s, ydot[f]);
      }
      float ytr[3] = {0.f, 0.f, 0.f};
#pragma unroll
      for (int j = 0; j < 5; ++j) {
#pragma unroll
        for (int kq = 0; kq < 5; ++kq) {
          float tp = shTP[(xc << 4) + (xnv[j] << 2) + xnv[kq]];
#pragma unroll
          for (int f = 0; f < 3; ++f) ytr[f] = fmaf(shanti[f*25 + j*5 + kq], tp, ytr[f]);
        }
      }
      float e0 = elu_f(ydot[0]), e1 = elu_f(ydot[1]), e2 = elu_f(ydot[2]);
      float e3 = elu_f(ytr[0]),  e4 = elu_f(ytr[1]),  e5 = elu_f(ytr[2]);
      unsigned* row = &shyw[wv][lane * 3];
      row[0] = pk16(e0, e1);
      row[1] = pk16(e2, e3);
      row[2] = pk16(e4, e5);
    }
    __syncthreads();

    // ---- stage 2: conv via WMMA + elu + accumulate ----
    if (active) {
      const unsigned* yw = &shyw[wv][0];
      v8u aw;
#pragma unroll
      for (int v = 0; v < 8; ++v) {
        int K = kbA + ((v < 4) ? 2*v : 16 + 2*(v - 4));   // even K-pair base for VGPR v
        aw[v] = (K < 30) ? yw[3*mrow + (K >> 1)] : 0u;
      }
      v16h Av = __builtin_bit_cast(v16h, aw);
      v8f c = {0.f, 0.f, 0.f, 0.f, 0.f, 0.f, 0.f, 0.f};
      c = __builtin_amdgcn_wmma_f32_16x16x32_f16(false, Av, false, Bv,
                                                 (short)0, c, false, false);
      // C layout: vgpr r -> (M=r, N=lane) for lanes 0-15; (M=8+r, N=lane-16) for 16-31
      if (mrow < 3) {
#pragma unroll
        for (int r = 0; r < 8; ++r) fsum += elu_f(c[r] + bb);
      }
    }
    __syncthreads();
  }

  // ---- deterministic per-block reduction of the 6 meaningful lanes per wave ----
  if (lane < 3)                 shred[wv][0][lane]      = fsum;  // M=0..7 half
  if (lane >= 16 && lane < 19)  shred[wv][1][lane - 16] = fsum;  // M=8..15 half
  __syncthreads();
  if (tid < 3) {
    float s = 0.f;
#pragma unroll
    for (int w = 0; w < 8; ++w) s += shred[w][0][tid] + shred[w][1][tid];
    partials[blockIdx.x * 3 + tid] = s;
  }
}

__global__ __launch_bounds__(256) void su2cnn_reduce_kernel(
    const float* __restrict__ partials,
    const float* __restrict__ dw,
    float* __restrict__ out,
    int nblocks, float invN)
{
  __shared__ float s0[256], s1[256], s2[256];
  int tid = threadIdx.x;
  float a0 = 0.f, a1 = 0.f, a2 = 0.f;
  for (int b = tid; b < nblocks; b += 256) {
    a0 += partials[3*b + 0];
    a1 += partials[3*b + 1];
    a2 += partials[3*b + 2];
  }
  s0[tid] = a0; s1[tid] = a1; s2[tid] = a2;
  __syncthreads();
  for (int off = 128; off > 0; off >>= 1) {
    if (tid < off) {
      s0[tid] += s0[tid + off];
      s1[tid] += s1[tid + off];
      s2[tid] += s2[tid + off];
    }
    __syncthreads();
  }
  if (tid == 0)
    out[0] = (s0[0]*dw[0] + s1[0]*dw[1] + s2[0]*dw[2]) * invN;
}

extern "C" void kernel_launch(void* const* d_in, const int* in_sizes, int n_in,
                              void* d_out, int out_size, void* d_ws, size_t ws_size,
                              hipStream_t stream) {
  (void)n_in; (void)out_size; (void)ws_size;
  const int*   x    = (const int*)d_in[0];
  const float* ikd  = (const float*)d_in[1];
  const float* ikt  = (const float*)d_in[2];
  const float* k0   = (const float*)d_in[3];
  const float* bias = (const float*)d_in[4];
  const float* dw   = (const float*)d_in[5];
  float* out      = (float*)d_out;
  float* partials = (float*)d_ws;   // 3 floats per block (12 KB)

  int n = in_sizes[0];              // N = 2^21, multiple of 16
  int tiles = n / 16;
  const int BLOCKS = 1024;
  int wavesTotal = BLOCKS * 8;
  int iters = (tiles + wavesTotal - 1) / wavesTotal;

  su2cnn_main_kernel<<<BLOCKS, 256, 0, stream>>>(x, ikd, ikt, k0, bias,
                                                 partials, n, tiles, iters, wavesTotal);
  su2cnn_reduce_kernel<<<1, 256, 0, stream>>>(partials, dw, out, BLOCKS,
                                              1.0f / (float)n);
}
